// EMAVectorQuantizer_74801150427612
// MI455X (gfx1250) — matrix operand order repaired and checked by reading, not compiled, not run on hardware
//
#include <hip/hip_runtime.h>
#include <hip/hip_bf16.h>

// ---------------------------------------------------------------------------
// EMA Vector Quantizer for MI455X (gfx1250, wave32, WMMA)
//
// Core: argmin_k ( ||e_k||^2 - 2 * x_n . e_k )  via v_wmma_f32_16x16x32_bf16.
// GEMM is 137 GFLOP; all other stages are HBM-bound (~130MB => ~6us @23.3TB/s).
// Block layout in the argmin kernel: 8 waves = 2 row-tiles x 4 code-splits,
// so each bf16 codebook fragment is reused by 2 waves via WGP$ and total L2
// codebook traffic is (N/32) * 8MB = 4GB instead of 8GB.
// ---------------------------------------------------------------------------

typedef __attribute__((ext_vector_type(16))) __bf16 v16bf;
typedef __attribute__((ext_vector_type(8)))  __bf16 v8bf;
typedef __attribute__((ext_vector_type(8)))  float  v8f;

#define EMA_DECAY    0.99f
#define ONE_MINUS    (1.0f - 0.99f)
#define COMMIT_W     0.25f
#define VQ_EPS       1e-5f

// ---------------- 1) f32 -> bf16 codebook conversion -----------------------
__global__ void vq_convert(const float* __restrict__ emb,
                           __bf16* __restrict__ embB, int KD) {
  int i = blockIdx.x * blockDim.x + threadIdx.x;
  if (i < KD) embB[i] = (__bf16)emb[i];   // RNE convert
}

// ---------------- 2) ||e_k||^2 (one wave per code) -------------------------
__global__ void vq_enorm(const float* __restrict__ emb,
                         float* __restrict__ enorm, int D, int K) {
  const int lane = threadIdx.x & 31;
  const int wave = threadIdx.x >> 5;
  const int k = blockIdx.x * (blockDim.x >> 5) + wave;
  if (k >= K) return;
  const float* e = emb + (size_t)k * D;
  float s = 0.f;
  for (int d = lane; d < D; d += 32) { float v = e[d]; s += v * v; }
  #pragma unroll
  for (int off = 16; off > 0; off >>= 1) s += __shfl_xor(s, off, 32);
  if (lane == 0) enorm[k] = s;
}

// ---------------- 3) init outputs + zero scalars ---------------------------
__global__ void vq_init(const float* __restrict__ ema_w,
                        const float* __restrict__ ema_cs,
                        float* __restrict__ new_w,
                        float* __restrict__ new_cs,
                        double* __restrict__ commit_sum,
                        float* __restrict__ cs_total,
                        int KD, int K) {
  int i = blockIdx.x * blockDim.x + threadIdx.x;
  if (i < KD) new_w[i]  = EMA_DECAY * ema_w[i];
  if (i < K)  new_cs[i] = EMA_DECAY * ema_cs[i];
  if (i == 0) { *commit_sum = 0.0; *cs_total = 0.f; }
}

// ---------------- 4) WMMA distance + argmin --------------------------------
// Block = 256 threads (8 waves) = 2 row-tiles x 4 code-splits over a 32-row
// feature tile held in LDS as bf16. Wave w: row-tile = w>>2, code-split = w&3
// sweeping code tiles t = split, split+4, ... Dual accumulators break the
// C-operand dependency chain; prefetch warms the next code tile's B rows.
__global__ void __launch_bounds__(256)
vq_argmin_wmma(const float* __restrict__ feat,
               const __bf16* __restrict__ embB,
               const float* __restrict__ enorm,
               int*  __restrict__ idx_out,
               float* __restrict__ tgt_out,
               int D, int K) {
  extern __shared__ char smem_raw[];
  __bf16* Atile  = (__bf16*)smem_raw;                         // 32 * D bf16
  float*  redMin = (float*)(smem_raw + (size_t)32 * D * 2);   // 32 rows x 4 splits
  int*    redIdx = (int*)(redMin + 32 * 4);

  const int tid     = threadIdx.x;
  const int lane    = tid & 31;
  const int wave    = tid >> 5;
  const int rgrp    = wave >> 2;     // 0..1 : which 16-row tile
  const int split   = wave & 3;      // 0..3 : code-tile stride offset
  const int rowbase = blockIdx.x * 32;

  // Cooperative load+convert of the 32xD f32 feature tile into bf16 LDS.
  const float* fbase = feat + (size_t)rowbase * D;
  for (int i = tid; i < 32 * D; i += blockDim.x)
    Atile[i] = (__bf16)fbase[i];
  __syncthreads();

  const int m  = lane & 15;   // A row (within tile) / B column / C column
  const int hi = lane >> 4;

  float best[8];
  int   bidx[8];
  #pragma unroll
  for (int v = 0; v < 8; ++v) { best[v] = 3.4e38f; bidx[v] = 0; }

  const int ntiles = K >> 4;
  const __bf16* arow = Atile + (size_t)(rgrp * 16 + m) * D;

  for (int t = split; t < ntiles; t += 4) {
    const int code0 = t << 4;
    const __bf16* brow = embB + (size_t)(code0 + m) * D;
    // Warm WGP$/GL2 for this wave's next code tile (t+4 => +64 rows).
    __builtin_prefetch(brow + (size_t)64 * D, 0, 0);
    v8f c0 = {};
    v8f c1 = {};
    for (int kb = 0; kb < D; kb += 64) {
      // Layout used here: A frag = 16 contiguous K at kb+16*hi for row m
      //             B frag = 16 contiguous K at kb+16*hi for column m
      v8bf a0lo = *(const v8bf*)(arow + kb + 16 * hi);
      v8bf a0hi = *(const v8bf*)(arow + kb + 8 + 16 * hi);
      v16bf a0 = __builtin_shufflevector(a0lo, a0hi,
                   0,1,2,3,4,5,6,7,8,9,10,11,12,13,14,15);
      v16bf b0 = *(const v16bf*)(brow + kb + 16 * hi);
      c0 = __builtin_amdgcn_wmma_f32_16x16x32_bf16(
             false, a0, false, b0, (short)0, c0, false, false);

      v8bf a1lo = *(const v8bf*)(arow + kb + 32 + 16 * hi);
      v8bf a1hi = *(const v8bf*)(arow + kb + 40 + 16 * hi);
      v16bf a1 = __builtin_shufflevector(a1lo, a1hi,
                   0,1,2,3,4,5,6,7,8,9,10,11,12,13,14,15);
      v16bf b1 = *(const v16bf*)(brow + kb + 32 + 16 * hi);
      c1 = __builtin_amdgcn_wmma_f32_16x16x32_bf16(
             false, a1, false, b1, (short)0, c1, false, false);
    }
    // C layout: VGPR v holds row (v + 8*hi), column (lane&15) => code0+m.
    const int   code = code0 + m;
    const float en   = enorm[code];
    #pragma unroll
    for (int v = 0; v < 8; ++v) {
      float s = en - 2.0f * (c0[v] + c1[v]);
      if (s < best[v]) { best[v] = s; bidx[v] = code; }
    }
  }

  // Cross-lane argmin within each 16-lane half (rows live per-half).
  #pragma unroll
  for (int v = 0; v < 8; ++v) {
    float bv = best[v]; int iv = bidx[v];
    #pragma unroll
    for (int off = 1; off < 16; off <<= 1) {
      float ob = __shfl_xor(bv, off, 32);
      int   oi = __shfl_xor(iv, off, 32);
      if (ob < bv || (ob == bv && oi < iv)) { bv = ob; iv = oi; }
    }
    best[v] = bv; bidx[v] = iv;
  }
  if (m == 0) {
    #pragma unroll
    for (int v = 0; v < 8; ++v) {
      int r = rgrp * 16 + v + 8 * hi;   // row within the 32-row tile
      redMin[r * 4 + split] = best[v];
      redIdx[r * 4 + split] = bidx[v];
    }
  }
  __syncthreads();
  if (tid < 32) {
    float bv = redMin[tid * 4];
    int   iv = redIdx[tid * 4];
    #pragma unroll
    for (int s4 = 1; s4 < 4; ++s4) {
      float ob = redMin[tid * 4 + s4];
      int   oi = redIdx[tid * 4 + s4];
      if (ob < bv || (ob == bv && oi < iv)) { bv = ob; iv = oi; }
    }
    const int row = rowbase + tid;
    idx_out[row] = iv;
    tgt_out[row] = (float)iv;
  }
}

// ---------------- 5) gather quantized + commitment + EMA scatter -----------
__global__ void __launch_bounds__(256)
vq_gather_scatter(const float* __restrict__ feat,
                  const float* __restrict__ emb,
                  const int* __restrict__ idx,
                  float* __restrict__ q_out,
                  float* __restrict__ new_w,
                  float* __restrict__ new_cs,
                  double* __restrict__ commit_sum,
                  int D) {
  const int n = blockIdx.x;
  const int k = idx[n];
  const float* x = feat + (size_t)n * D;
  const float* e = emb  + (size_t)k * D;
  float* q = q_out + (size_t)n * D;
  float* w = new_w + (size_t)k * D;

  float local = 0.f;
  for (int d = threadIdx.x; d < D; d += blockDim.x) {
    float xv = x[d], ev = e[d];
    q[d] = ev;                               // quantized_st == quantized (values)
    float df = xv - ev;
    local += df * df;
    atomicAdd(&w[d], ONE_MINUS * xv);        // (1-decay) * embed_sum scatter
  }
  __shared__ float sred[256];
  sred[threadIdx.x] = local;
  __syncthreads();
  for (int s = 128; s > 0; s >>= 1) {
    if (threadIdx.x < s) sred[threadIdx.x] += sred[threadIdx.x + s];
    __syncthreads();
  }
  if (threadIdx.x == 0) {
    atomicAdd(commit_sum, (double)sred[0]);
    atomicAdd(&new_cs[k], ONE_MINUS);        // (1-decay) * cluster_size scatter
  }
}

// ---------------- 6) sum(new_cs) (single block => deterministic) -----------
__global__ void vq_cs_total(const float* __restrict__ new_cs,
                            float* __restrict__ cs_total, int K) {
  __shared__ float sred[256];
  float s = 0.f;
  for (int i = threadIdx.x; i < K; i += 256) s += new_cs[i];
  sred[threadIdx.x] = s;
  __syncthreads();
  for (int st = 128; st > 0; st >>= 1) {
    if (threadIdx.x < st) sred[threadIdx.x] += sred[threadIdx.x + st];
    __syncthreads();
  }
  if (threadIdx.x == 0) *cs_total = sred[0];
}

// ---------------- 7) new_embeddings + loss ---------------------------------
__global__ void vq_final(const float* __restrict__ new_w,
                         const float* __restrict__ new_cs,
                         float* __restrict__ new_emb,
                         const float* __restrict__ cs_total,
                         const double* __restrict__ commit_sum,
                         float* __restrict__ loss_out,
                         int KD, int D, int K, long long BTD) {
  int i = blockIdx.x * blockDim.x + threadIdx.x;
  if (i < KD) {
    int k = i / D;
    float total = *cs_total + (float)K * VQ_EPS;
    float csn   = (new_cs[k] + VQ_EPS) / total;
    new_emb[i]  = new_w[i] / csn;
  }
  if (i == 0)
    *loss_out = COMMIT_W * (float)(*commit_sum / (double)BTD);
}

// ---------------------------------------------------------------------------
extern "C" void kernel_launch(void* const* d_in, const int* in_sizes, int n_in,
                              void* d_out, int out_size, void* d_ws, size_t ws_size,
                              hipStream_t stream) {
  const float* feat   = (const float*)d_in[0];   // [B,T,D] f32
  const float* emb    = (const float*)d_in[1];   // [K,D]   f32
  const float* ema_cs = (const float*)d_in[2];   // [K]     f32
  const float* ema_w  = (const float*)d_in[3];   // [K,D]   f32

  const long long BTD = (long long)in_sizes[0];
  const int KD = in_sizes[1];
  const int K  = in_sizes[2];
  const int D  = KD / K;                         // 512
  const int N  = (int)(BTD / D);                 // 16384

  // Output layout: quantized_st | targets | loss | new_embeddings | new_cs | new_w
  float* out     = (float*)d_out;
  float* q_out   = out;
  float* tgt_out = out + BTD;
  float* loss    = tgt_out + N;
  float* newemb  = loss + 1;
  float* newcs   = newemb + KD;
  float* neww    = newcs + K;

  // Workspace: bf16 codebook | enorm | indices | scalars (~8.5 MB)
  char* ws = (char*)d_ws;
  size_t o = 0;
  __bf16* embB = (__bf16*)(ws + o);  o += (size_t)KD * 2;  o = (o + 255) & ~(size_t)255;
  float*  en   = (float*)(ws + o);   o += (size_t)K * 4;   o = (o + 255) & ~(size_t)255;
  int*    idx  = (int*)(ws + o);     o += (size_t)N * 4;   o = (o + 255) & ~(size_t)255;
  double* csum = (double*)(ws + o);  o += 8;
  float*  ctot = (float*)(ws + o);

  vq_convert<<<(KD + 255) / 256, 256, 0, stream>>>(emb, embB, KD);
  vq_enorm<<<(K + 7) / 8, 256, 0, stream>>>(emb, en, D, K);
  vq_init<<<(KD + 255) / 256, 256, 0, stream>>>(ema_w, ema_cs, neww, newcs,
                                                csum, ctot, KD, K);
  size_t smem = (size_t)32 * D * 2 + (size_t)32 * 4 * 8;
  vq_argmin_wmma<<<N / 32, 256, smem, stream>>>(feat, embB, en, idx, tgt_out, D, K);
  vq_gather_scatter<<<N, 256, 0, stream>>>(feat, emb, idx, q_out, neww, newcs,
                                           csum, D);
  vq_cs_total<<<1, 256, 0, stream>>>(newcs, ctot, K);
  vq_final<<<(KD + 255) / 256, 256, 0, stream>>>(neww, newcs, newemb, ctot,
                                                 csum, loss, KD, D, K, BTD);
}
